// SSM_11424613007519
// MI455X (gfx1250) — compile-verified, matrix operand order checked by reading, not run on hardware
//
#include <hip/hip_runtime.h>
#include <hip/hip_bf16.h>

// Problem constants (from reference): B=4, T=4096, D=1024, KSIZE=128
#define BB   4
#define TT_  4096
#define DD   1024
#define KF   128
#define MM   (BB * TT_)          // 16384 rows for the GEMMs

typedef __attribute__((ext_vector_type(16))) __bf16 v16bf;
typedef __attribute__((ext_vector_type(8)))  __bf16 v8bf;
typedef __attribute__((ext_vector_type(8)))  float  v8f;

#if __has_builtin(__builtin_amdgcn_global_load_async_to_lds_b128) && \
    __has_builtin(__builtin_amdgcn_s_wait_asynccnt)
#define HAVE_ASYNC_LDS 1
// per clang diagnostic: params are pointers to int __vector_size__(16) in
// AS(1) (global, printed as __device__) and AS(3) (LDS)
typedef int vi4 __attribute__((vector_size(4 * sizeof(int))));
typedef __attribute__((address_space(1))) vi4* gv4p;
typedef __attribute__((address_space(3))) vi4* lv4p;
#else
#define HAVE_ASYNC_LDS 0
#endif

__device__ __forceinline__ unsigned short f2bf(float f) {
    unsigned int u = __float_as_uint(f);
    unsigned int r = u + 0x7FFFu + ((u >> 16) & 1u);   // round-to-nearest-even
    return (unsigned short)(r >> 16);
}
__device__ __forceinline__ float bf2f(unsigned short s) {
    return __uint_as_float(((unsigned int)s) << 16);
}

// ---------------------------------------------------------------------------
// Kernel 1: FIR kernel generation + tanh(beta).  fir stored [k][d] (transposed)
// ---------------------------------------------------------------------------
__global__ void fir_prep(const float* __restrict__ rg, const float* __restrict__ rb,
                         const float* __restrict__ Cc,
                         float* __restrict__ fir, float* __restrict__ tb) {
    int d = blockIdx.x * blockDim.x + threadIdx.x;
    if (d >= DD) return;
    float g  = rg[d];
    float sp = (g > 20.f) ? g : log1pf(expf(g));       // softplus
    float gamma = sp + 1e-4f;
    float c = Cc[d];
    float sum = 0.f;
    for (int k = 0; k < KF; ++k) {
        float v = c * expf(-gamma * (float)k);
        sum += v * v;
    }
    float inv = 1.f / (sqrtf(sum) + 1e-6f);
    for (int k = 0; k < KF; ++k) {
        fir[k * DD + d] = c * expf(-gamma * (float)k) * inv;
    }
    tb[d] = tanhf(rb[d]);
}

// ---------------------------------------------------------------------------
// Kernel 2: f32 -> bf16 cast (vectorized x4)
// ---------------------------------------------------------------------------
__global__ void cast_bf16(const float* __restrict__ in, unsigned short* __restrict__ out, int n) {
    int i = (blockIdx.x * blockDim.x + threadIdx.x) * 4;
    if (i + 3 >= n) return;
    float4 f = *(const float4*)(in + i);
    union { unsigned short s[4]; uint2 u; } o;
    o.s[0] = f2bf(f.x); o.s[1] = f2bf(f.y); o.s[2] = f2bf(f.z); o.s[3] = f2bf(f.w);
    *(uint2*)(out + i) = o.u;
}

// ---------------------------------------------------------------------------
// Kernel 3/5: WMMA bf16 GEMM:  out[m,n] = act( sum_k A[m,k]*Bm[n,k] + bias[n] )
//   MODE 0: sigmoid -> bf16 store (dynamic_scale)
//   MODE 1: identity -> f32 store (final output)
// Block: 256 threads = 8 waves; tile 128M x 64N; K-step 32.
// B tile (64N x 32K, shared by all 8 waves) is double-buffered in LDS via
// GLOBAL_LOAD_ASYNC_TO_LDS_B128 (ASYNCcnt); padded row stride (40 bf16 = 80 B)
// makes the ds_load_b128 fragment reads bank-conflict-free.
// A fragments stream straight from global (per-wave rows).
// ---------------------------------------------------------------------------
template <int MODE>
__global__ __launch_bounds__(256)
void gemm_bias_act(const __bf16* __restrict__ A, const __bf16* __restrict__ Bm,
                   const float* __restrict__ bias, void* __restrict__ out) {
    constexpr int ROWB = 40;                 // padded LDS row stride (bf16 elems)
    __shared__ __bf16 bs[2][64 * ROWB];      // 2 x 5120 B

    const int tid  = threadIdx.x;
    const int lane = tid & 31;
    const int wave = tid >> 5;
    const int g    = lane >> 4;              // half-wave selector
    const int l    = lane & 15;
    const int mBase = blockIdx.y * 128 + wave * 16;
    const int nBase = blockIdx.x * 64;

    const __bf16* arow = A + (size_t)(mBase + l) * DD;

    // cooperative B-tile copy addressing: thread -> (row, 16B chunk)
    const int brow = tid >> 2;               // 0..63  (N within tile)
    const int bchk = tid & 3;                // 0..3   (chunk of 8 bf16 along K)
    const __bf16* bsrc = Bm + (size_t)(nBase + brow) * DD + bchk * 8;
    __bf16* bdst[2] = { &bs[0][brow * ROWB + bchk * 8],
                        &bs[1][brow * ROWB + bchk * 8] };

    v8f z8 = {0.f,0.f,0.f,0.f,0.f,0.f,0.f,0.f};
    v8f acc[4] = {z8, z8, z8, z8};

    // prologue: prefetch K-step 0 into buffer 0
#if HAVE_ASYNC_LDS
    __builtin_amdgcn_global_load_async_to_lds_b128(
        (gv4p)(void*)(bsrc), (lv4p)(void*)(bdst[0]), 0, 0);
#else
    *(uint4*)bdst[0] = *(const uint4*)(bsrc);
#endif

    for (int k0 = 0, buf = 0; k0 < DD; k0 += 32, buf ^= 1) {
#if HAVE_ASYNC_LDS
        __builtin_amdgcn_s_wait_asynccnt(0);   // own chunk landed in LDS
#endif
        __syncthreads();                        // publish tile; prev reads done
        if (k0 + 32 < DD) {
#if HAVE_ASYNC_LDS
            __builtin_amdgcn_global_load_async_to_lds_b128(
                (gv4p)(void*)(bsrc + k0 + 32), (lv4p)(void*)(bdst[buf ^ 1]), 0, 0);
#else
            *(uint4*)bdst[buf ^ 1] = *(const uint4*)(bsrc + k0 + 32);
#endif
        }

        union { v16bf v; v8bf h[2]; } af;
        af.h[0] = *(const v8bf*)(arow + k0 + g * 8);
        af.h[1] = *(const v8bf*)(arow + k0 + 16 + g * 8);
        const __bf16* bb = bs[buf];
#pragma unroll
        for (int nt = 0; nt < 4; ++nt) {
            const __bf16* brp = bb + (nt * 16 + l) * ROWB;
            union { v16bf v; v8bf h[2]; } bfr;
            bfr.h[0] = *(const v8bf*)(brp + g * 8);
            bfr.h[1] = *(const v8bf*)(brp + 16 + g * 8);
            acc[nt] = __builtin_amdgcn_wmma_f32_16x16x32_bf16(
                false, af.v, false, bfr.v, (short)0, acc[nt], false, false);
        }
    }

    // C/D layout: lane holds column n = l, rows m = g*8 + r (r = VGPR index)
#pragma unroll
    for (int nt = 0; nt < 4; ++nt) {
        int n = nBase + nt * 16 + l;
        float bv = bias[n];
#pragma unroll
        for (int r = 0; r < 8; ++r) {
            int m = mBase + g * 8 + r;
            float v = acc[nt][r] + bv;
            if (MODE == 0) {
                v = 1.f / (1.f + expf(-v));
                ((unsigned short*)out)[(size_t)m * DD + n] = f2bf(v);
            } else {
                ((float*)out)[(size_t)m * DD + n] = v;
            }
        }
    }
}

// ---------------------------------------------------------------------------
// Kernel 4: causal depthwise conv + gate + residual, LDS-tiled.
//   y[b,t,d] = tanh(beta[d]) * sum_k fir[d,k] * x[b, t+k-127, d]   (XLA xcorr)
//   z[b,t,d] = y * scale + x   -> bf16
// Tile: 64 t x 32 d per block; LDS: x tile (191x32, 24.4KB) + fir tile (128x32, 16KB)
// ---------------------------------------------------------------------------
__global__ __launch_bounds__(256)
void conv_gate(const float* __restrict__ x, const float* __restrict__ fir,
               const float* __restrict__ tb, const unsigned short* __restrict__ scale,
               unsigned short* __restrict__ zbf) {
    constexpr int CT = 64, CD = 32;
    __shared__ float xs[(CT + KF - 1) * CD];   // rows t0-127 .. t0+63
    __shared__ float fs[KF * CD];              // fir[k][dLocal]

    const int b  = blockIdx.z;
    const int t0 = blockIdx.x * CT;
    const int d0 = blockIdx.y * CD;
    const int tid = threadIdx.x;
    const size_t base = (size_t)b * TT_ * DD;

    for (int idx = tid; idx < KF * CD; idx += 256) {
        int k = idx >> 5, dc = idx & 31;
        fs[idx] = fir[k * DD + d0 + dc];
    }
    for (int idx = tid; idx < (CT + KF - 1) * CD; idx += 256) {
        int row = idx >> 5, dc = idx & 31;
        int tg = t0 - (KF - 1) + row;
        xs[idx] = (tg >= 0) ? x[base + (size_t)tg * DD + d0 + dc] : 0.f;
    }
    __syncthreads();

    const int dl   = tid & 31;       // lane == d offset: conflict-free LDS
    const int tgrp = tid >> 5;       // 0..7
    const int d    = d0 + dl;
    const float tbv = tb[d];

#pragma unroll
    for (int i = 0; i < 8; ++i) {
        int tl = tgrp * 8 + i;
        float acc = 0.f;
#pragma unroll 8
        for (int k = 0; k < KF; ++k)
            acc += fs[k * CD + dl] * xs[(tl + k) * CD + dl];
        int t = t0 + tl;
        size_t off = base + (size_t)t * DD + d;
        float sc = bf2f(scale[off]);
        float z  = acc * tbv * sc + xs[(tl + KF - 1) * CD + dl];  // + residual x
        zbf[off] = f2bf(z);
    }
}

// ---------------------------------------------------------------------------
// Launch
// ---------------------------------------------------------------------------
extern "C" void kernel_launch(void* const* d_in, const int* in_sizes, int n_in,
                              void* d_out, int out_size, void* d_ws, size_t ws_size,
                              hipStream_t stream) {
    (void)in_sizes; (void)n_in; (void)out_size; (void)ws_size;
    const float* x  = (const float*)d_in[0];
    const float* rg = (const float*)d_in[1];
    const float* rb = (const float*)d_in[2];
    const float* Cc = (const float*)d_in[3];
    const float* Wd = (const float*)d_in[4];
    const float* bd = (const float*)d_in[5];
    const float* Wo = (const float*)d_in[6];
    const float* bo = (const float*)d_in[7];

    char* ws = (char*)d_ws;
    const size_t MD = (size_t)MM * DD;           // 16M elements
    unsigned short* xbf   = (unsigned short*)(ws);                        // 32 MB
    unsigned short* zbf   = (unsigned short*)(ws + MD * 2);               // 32 MB
    unsigned short* scbf  = (unsigned short*)(ws + MD * 4);               // 32 MB
    unsigned short* wdbf  = (unsigned short*)(ws + MD * 6);               //  2 MB
    unsigned short* wobf  = (unsigned short*)(ws + MD * 6 + (size_t)DD * DD * 2);
    float*          fir   = (float*)(ws + MD * 6 + (size_t)DD * DD * 4);  // 512 KB
    float*          tb    = (float*)((char*)fir + (size_t)DD * KF * 4);   //   4 KB

    // 1) FIR + tanh(beta)
    fir_prep<<<(DD + 255) / 256, 256, 0, stream>>>(rg, rb, Cc, fir, tb);

    // 2) bf16 casts
    cast_bf16<<<(int)(MD / 4 + 255) / 256, 256, 0, stream>>>(x,  xbf,  (int)MD);
    cast_bf16<<<(DD * DD / 4 + 255) / 256, 256, 0, stream>>>(Wd, wdbf, DD * DD);
    cast_bf16<<<(DD * DD / 4 + 255) / 256, 256, 0, stream>>>(Wo, wobf, DD * DD);

    // 3) dynamic_scale = sigmoid(x @ Wd^T + bd)  (bf16 out)
    dim3 gGemm(DD / 64, MM / 128);
    gemm_bias_act<0><<<gGemm, 256, 0, stream>>>(
        (const __bf16*)xbf, (const __bf16*)wdbf, bd, (void*)scbf);

    // 4) depthwise conv + gate + residual -> z (bf16)
    dim3 gConv(TT_ / 64, DD / 32, BB);
    conv_gate<<<gConv, 256, 0, stream>>>(x, fir, tb, scbf, zbf);

    // 5) out = z @ Wo^T + bo (f32 out)
    gemm_bias_act<1><<<gGemm, 256, 0, stream>>>(
        (const __bf16*)zbf, (const __bf16*)wobf, bo, d_out);
}
